// CausalSelfAttention_10299331576322
// MI455X (gfx1250) — matrix operand order repaired and checked by reading, not compiled, not run on hardware
//
#include <hip/hip_runtime.h>
#include <hip/hip_bf16.h>

// ---------------------------------------------------------------------------
// Causal self-attention (FIRE additive mask) on MI455X / gfx1250.
// All four matmuls run on v_wmma_f32_16x16x32_bf16 (f32 accumulate).
// K-tiles in flash attention are staged via global_load_async_to_lds_b128.
// B=4, T=2048, C=768, H=12, hd=64.
// ---------------------------------------------------------------------------

#define BATCH 4
#define T_SEQ 2048
#define C_EMB 768
#define N_HEAD 12
#define HD 64
#define C3 (3 * C_EMB)     // 2304
#define BT (BATCH * T_SEQ) // 8192

typedef __attribute__((ext_vector_type(16))) __bf16 v16bf;
typedef __attribute__((ext_vector_type(8)))  float  v8f;
typedef __attribute__((ext_vector_type(4)))  unsigned int u32x4;

union Frag {          // 8 VGPRs viewed as 16 bf16 (WMMA A/B operand)
    u32x4 u[2];
    v16bf v;
};

__device__ __forceinline__ v8f wmma_bf16(v16bf a, v16bf b, v8f c) {
    // (neg_a, A, neg_b, B, c_mod, C, reuse_a, reuse_b)
    return __builtin_amdgcn_wmma_f32_16x16x32_bf16(false, a, false, b,
                                                   (short)0, c, false, false);
}

__device__ __forceinline__ unsigned lds_off(const void* p) {
    // __shared__ object -> 32-bit LDS byte offset (low bits of generic addr)
    return (unsigned)(unsigned long long)p;
}

// ---------------------------------------------------------------------------
// f32 -> bf16 conversion
// ---------------------------------------------------------------------------
__global__ void f32_to_bf16_kernel(const float* __restrict__ src,
                                   __bf16* __restrict__ dst, int n) {
    int i = blockIdx.x * blockDim.x + threadIdx.x;
    if (i < n) dst[i] = (__bf16)src[i];
}

// ---------------------------------------------------------------------------
// GEMM: out[M,N] = A[M,K](bf16) * B[K,N](bf16) + bias[N](f32)
// Block tile 256x64, 256 threads = 8 waves, each wave owns a 32x64 strip
// (two 16-row A strips sharing the four B fragments -> 8 WMMAs per sync).
// ---------------------------------------------------------------------------
template <bool OUT_BF16>
__global__ __launch_bounds__(256) void gemm_bias_kernel(
    const __bf16* __restrict__ A, const __bf16* __restrict__ Bm,
    const float* __restrict__ bias, void* __restrict__ outp,
    int M, int N, int K)
{
    __shared__ __align__(16) __bf16 As[256][40];  // 256x32 tile (+16B pad)
    __shared__ __align__(16) __bf16 Bt[64][40];   // 32x64 tile, transposed

    const int tid  = threadIdx.x;
    const int wave = tid >> 5;
    const int lane = tid & 31;
    const int half = lane >> 4;
    const int lr   = lane & 15;
    const int m0   = blockIdx.y * 256;
    const int n0   = blockIdx.x * 64;

    v8f acc[8] = {v8f{}, v8f{}, v8f{}, v8f{}, v8f{}, v8f{}, v8f{}, v8f{}};

    for (int k0 = 0; k0 < K; k0 += 32) {
        // ---- stage A: 256x32 bf16 = 1024 x b128 loads
        #pragma unroll
        for (int i = 0; i < 4; ++i) {
            int idx = i * 256 + tid;
            int r = idx >> 2, cv = (idx & 3) * 8;
            *(u32x4*)&As[r][cv] =
                *(const u32x4*)&A[(size_t)(m0 + r) * K + k0 + cv];
        }
        // ---- stage B transposed: Bt[n][k] so fragments are contiguous
        {
            int r = tid >> 3, cv = (tid & 7) * 8;
            u32x4 raw = *(const u32x4*)&Bm[(size_t)(k0 + r) * N + n0 + cv];
            const __bf16* e = (const __bf16*)&raw;
            #pragma unroll
            for (int j = 0; j < 8; ++j) Bt[cv + j][r] = e[j];
        }
        __syncthreads();

        // ---- two A fragments (rows wave*32 .. wave*32+31)
        Frag af0, af1;
        af0.u[0] = *(const u32x4*)&As[wave * 32 + lr][half * 8];
        af0.u[1] = *(const u32x4*)&As[wave * 32 + lr][16 + half * 8];
        af1.u[0] = *(const u32x4*)&As[wave * 32 + 16 + lr][half * 8];
        af1.u[1] = *(const u32x4*)&As[wave * 32 + 16 + lr][16 + half * 8];

        #pragma unroll
        for (int nt = 0; nt < 4; ++nt) {
            Frag bf_;
            bf_.u[0] = *(const u32x4*)&Bt[nt * 16 + lr][half * 8];
            bf_.u[1] = *(const u32x4*)&Bt[nt * 16 + lr][16 + half * 8];
            acc[nt]     = wmma_bf16(af0.v, bf_.v, acc[nt]);
            acc[nt + 4] = wmma_bf16(af1.v, bf_.v, acc[nt + 4]);
        }
        __syncthreads();
    }

    // ---- epilogue (C layout: VGPR r holds rows r / r+8 by lane half)
    #pragma unroll
    for (int nt = 0; nt < 4; ++nt) {
        int col = n0 + nt * 16 + lr;
        float bv = bias[col];
        #pragma unroll
        for (int strip = 0; strip < 2; ++strip) {
            #pragma unroll
            for (int r = 0; r < 8; ++r) {
                int row = m0 + wave * 32 + strip * 16 + r + half * 8;
                float vv = acc[nt + strip * 4][r] + bv;
                if (OUT_BF16)
                    ((__bf16*)outp)[(size_t)row * N + col] = (__bf16)vv;
                else
                    ((float*)outp)[(size_t)row * N + col] = vv;
            }
        }
    }
}

// ---------------------------------------------------------------------------
// Flash attention: one block per (64 query rows, head, batch).
// 128 threads = 4 waves; wave w owns query rows [qt*64 + w*16, +16).
// K tile staged by async-to-LDS DMA; V tile transposed through VGPRs.
// ---------------------------------------------------------------------------
__global__ __launch_bounds__(128) void flash_attn_kernel(
    const __bf16* __restrict__ qkv,   // [BT, 2304] bf16 (q|k|v)
    const float*  __restrict__ mask,  // [H, T, T] f32
    __bf16* __restrict__ y)           // [BT, 768] bf16
{
    __shared__ __align__(16) __bf16 Ks[64][64];      // K tile  [key][d]
    __shared__ __align__(16) __bf16 Vt[64][64];      // V tile  [d][key]
    __shared__ __align__(16) __bf16 Ps[4][16][64];   // P tiles per wave

    const int tid  = threadIdx.x;
    const int wave = tid >> 5;
    const int lane = tid & 31;
    const int half = lane >> 4;
    const int lr   = lane & 15;
    const int qt = blockIdx.x, h = blockIdx.y, b = blockIdx.z;

    // ---- Q fragments (A layout), held in registers for the whole block
    const int qrow = qt * 64 + wave * 16 + lr;
    const __bf16* qbase = qkv + ((size_t)(b * T_SEQ) + qrow) * C3 + h * HD;
    Frag qf[2];
    #pragma unroll
    for (int kk = 0; kk < 2; ++kk) {
        qf[kk].u[0] = *(const u32x4*)&qbase[kk * 32 + half * 8];
        qf[kk].u[1] = *(const u32x4*)&qbase[kk * 32 + 16 + half * 8];
    }

    v8f O[4] = {v8f{}, v8f{}, v8f{}, v8f{}};
    float m_i[8], l_i[8];
    #pragma unroll
    for (int r = 0; r < 8; ++r) { m_i[r] = -3.0e38f; l_i[r] = 0.0f; }
    const float scale = 0.125f; // 1/sqrt(64)

    for (int kt = 0; kt <= qt; ++kt) {          // causal: mask handles diagonal
        // ---- K tile: async DMA straight into LDS (ASYNCcnt-tracked)
        #pragma unroll
        for (int i = 0; i < 4; ++i) {
            int t = i * 128 + tid;               // 512 x b128 transfers
            int key = t >> 3, cv = (t & 7) * 8;
            unsigned voff = (unsigned)(((((b * T_SEQ) + kt * 64 + key) * C3)
                                        + C_EMB + h * HD + cv) * 2);
            unsigned ldst = lds_off(&Ks[key][cv]);
            asm volatile("global_load_async_to_lds_b128 %0, %1, %2"
                         :: "v"(ldst), "v"(voff), "s"(qkv) : "memory");
        }
        // ---- V tile: load through VGPRs, store transposed
        #pragma unroll
        for (int i = 0; i < 4; ++i) {
            int idx = i * 128 + tid;
            int key = idx >> 3, cv = (idx & 7) * 8;
            size_t rbase = ((size_t)(b * T_SEQ) + kt * 64 + key) * C3 + h * HD;
            u32x4 raw = *(const u32x4*)&qkv[rbase + 2 * C_EMB + cv];
            const __bf16* e = (const __bf16*)&raw;
            #pragma unroll
            for (int j = 0; j < 8; ++j) Vt[cv + j][key] = e[j];
        }
        asm volatile("s_wait_asynccnt 0" ::: "memory");
        __syncthreads();

        // ---- S = Q K^T (B-fragment: column = key, contiguous d from Ks row)
        v8f S[4];
        #pragma unroll
        for (int nt = 0; nt < 4; ++nt) {
            v8f acc = {};
            #pragma unroll
            for (int kk = 0; kk < 2; ++kk) {
                Frag kb;
                kb.u[0] = *(const u32x4*)&Ks[nt * 16 + lr][kk * 32 + half * 8];
                kb.u[1] = *(const u32x4*)&Ks[nt * 16 + lr][kk * 32 + 16 + half * 8];
                acc = wmma_bf16(qf[kk].v, kb.v, acc);
            }
            S[nt] = acc;
        }

        // ---- scale + FIRE mask + online softmax (rows r, r+8 per lane half)
        float mnew[8];
        #pragma unroll
        for (int r = 0; r < 8; ++r) {
            int row = qt * 64 + wave * 16 + r + half * 8;
            const float* mrow = mask + ((size_t)h * T_SEQ + row) * T_SEQ;
            float mx = m_i[r];
            #pragma unroll
            for (int nt = 0; nt < 4; ++nt) {
                int col = kt * 64 + nt * 16 + lr;
                float s = S[nt][r] * scale + mrow[col];
                S[nt][r] = s;
                mx = fmaxf(mx, s);
            }
            #pragma unroll
            for (int d = 1; d < 16; d <<= 1)
                mx = fmaxf(mx, __shfl_xor(mx, d, 32));
            mnew[r] = mx;
        }
        #pragma unroll
        for (int r = 0; r < 8; ++r) {
            float corr = __expf(m_i[r] - mnew[r]);
            float rs = 0.0f;
            #pragma unroll
            for (int nt = 0; nt < 4; ++nt) {
                float p = __expf(S[nt][r] - mnew[r]);
                S[nt][r] = p;
                rs += p;
            }
            #pragma unroll
            for (int d = 1; d < 16; d <<= 1) rs += __shfl_xor(rs, d, 32);
            l_i[r] = l_i[r] * corr + rs;
            m_i[r] = mnew[r];
            #pragma unroll
            for (int nt = 0; nt < 4; ++nt) O[nt][r] *= corr;
        }

        // ---- P: C-layout -> A-layout via wave-private LDS round trip
        #pragma unroll
        for (int nt = 0; nt < 4; ++nt)
            #pragma unroll
            for (int r = 0; r < 8; ++r)
                Ps[wave][r + half * 8][nt * 16 + lr] = (__bf16)S[nt][r];
        asm volatile("s_wait_dscnt 0" ::: "memory");

        Frag pf[2];
        #pragma unroll
        for (int kk = 0; kk < 2; ++kk) {
            pf[kk].u[0] = *(const u32x4*)&Ps[wave][lr][kk * 32 + half * 8];
            pf[kk].u[1] = *(const u32x4*)&Ps[wave][lr][kk * 32 + 16 + half * 8];
        }

        // ---- O += P V (B-fragment: column = d, contiguous keys from Vt row)
        #pragma unroll
        for (int dt = 0; dt < 4; ++dt) {
            #pragma unroll
            for (int kk = 0; kk < 2; ++kk) {
                Frag vb;
                vb.u[0] = *(const u32x4*)&Vt[dt * 16 + lr][kk * 32 + half * 8];
                vb.u[1] = *(const u32x4*)&Vt[dt * 16 + lr][kk * 32 + 16 + half * 8];
                O[dt] = wmma_bf16(pf[kk].v, vb.v, O[dt]);
            }
        }
        __syncthreads();
    }

    // ---- normalize and emit y (bf16, [BT, C] layout for the output GEMM)
    #pragma unroll
    for (int dt = 0; dt < 4; ++dt) {
        #pragma unroll
        for (int r = 0; r < 8; ++r) {
            int row = qt * 64 + wave * 16 + r + half * 8;
            float vv = O[dt][r] / l_i[r];
            y[((size_t)(b * T_SEQ) + row) * C_EMB + h * HD + dt * 16 + lr] =
                (__bf16)vv;
        }
    }
}

// ---------------------------------------------------------------------------
// Host-side launch
// ---------------------------------------------------------------------------
extern "C" void kernel_launch(void* const* d_in, const int* in_sizes, int n_in,
                              void* d_out, int out_size, void* d_ws, size_t ws_size,
                              hipStream_t stream) {
    (void)in_sizes; (void)n_in; (void)out_size; (void)ws_size;

    const float* x      = (const float*)d_in[0];   // [B,T,C]
    const float* mask   = (const float*)d_in[1];   // [1,H,T,T]
    const float* W_attn = (const float*)d_in[2];   // [C,3C]
    const float* b_attn = (const float*)d_in[3];   // [3C]
    const float* W_proj = (const float*)d_in[4];   // [C,C]
    const float* b_proj = (const float*)d_in[5];   // [C]
    float* out = (float*)d_out;

    char* ws = (char*)d_ws;
    size_t off = 0;
    __bf16* xb  = (__bf16*)(ws + off); off += (size_t)BT * C_EMB * 2;
    __bf16* wA  = (__bf16*)(ws + off); off += (size_t)C_EMB * C3 * 2;
    __bf16* wP  = (__bf16*)(ws + off); off += (size_t)C_EMB * C_EMB * 2;
    __bf16* qkv = (__bf16*)(ws + off); off += (size_t)BT * C3 * 2;
    __bf16* yb  = (__bf16*)(ws + off); off += (size_t)BT * C_EMB * 2;

    // f32 -> bf16 staging
    {
        int n = BT * C_EMB;
        f32_to_bf16_kernel<<<(n + 255) / 256, 256, 0, stream>>>(x, xb, n);
    }
    {
        int n = C_EMB * C3;
        f32_to_bf16_kernel<<<(n + 255) / 256, 256, 0, stream>>>(W_attn, wA, n);
    }
    {
        int n = C_EMB * C_EMB;
        f32_to_bf16_kernel<<<(n + 255) / 256, 256, 0, stream>>>(W_proj, wP, n);
    }

    // qkv = x @ W_attn + b_attn   (bf16 out)
    gemm_bias_kernel<true><<<dim3(C3 / 64, BT / 256), 256, 0, stream>>>(
        xb, wA, b_attn, qkv, BT, C3, C_EMB);

    // flash attention with FIRE causal mask
    flash_attn_kernel<<<dim3(T_SEQ / 64, N_HEAD, BATCH), 128, 0, stream>>>(
        qkv, mask, yb);

    // out = y @ W_proj + b_proj   (f32 out)
    gemm_bias_kernel<false><<<dim3(C_EMB / 64, BT / 256), 256, 0, stream>>>(
        yb, wP, b_proj, out, BT, C_EMB, C_EMB);
}
